// Model_5446018531919
// MI455X (gfx1250) — matrix operand order repaired and checked by reading, not compiled, not run on hardware
//
#include <hip/hip_runtime.h>
#include <hip/hip_bf16.h>
#include <math.h>

#define NN 100000
#define NE 1600000
#define NG 64
#define NH 2

typedef __attribute__((ext_vector_type(16))) _Float16 v16h;
typedef __attribute__((ext_vector_type(8)))  _Float16 v8h;
typedef __attribute__((ext_vector_type(8)))  float    v8f;

__device__ __forceinline__ float leaky(float x, float s){ return x > 0.f ? x : s * x; }

// segment-max via monotone bit-pattern atomics (classic float max trick)
__device__ __forceinline__ void atomicMaxF(float* addr, float v){
  if (v >= 0.f) atomicMax((int*)addr, __float_as_int(v));
  else          atomicMin((unsigned int*)addr, __float_as_uint(v));
}

__device__ __forceinline__ v16h cat16(v8h lo, v8h hi){
  return __builtin_shufflevector(lo, hi, 0,1,2,3,4,5,6,7,8,9,10,11,12,13,14,15);
}

// ---------------- utility kernels ----------------
__global__ void k_fill_f32(float* __restrict__ p, float v, long long n){
  long long i = (long long)blockIdx.x * blockDim.x + threadIdx.x;
  if (i < n) p[i] = v;
}

__global__ void k_cast_f16(_Float16* __restrict__ dst, const float* __restrict__ src, long long n){
  long long i = (long long)blockIdx.x * blockDim.x + threadIdx.x;
  if (i < n) dst[i] = (_Float16)src[i];
}

// pack weights transposed+cast: src [H][Din][Dout] f32 -> dst [H][Dout][Din] f16
// (per-lane B fragment then = two runs of 8 contiguous halves, 16 apart -> two b128 loads)
__global__ void k_pack_w(_Float16* __restrict__ dst, const float* __restrict__ src,
                         int Din, int Dout, int total){
  int i = blockIdx.x * blockDim.x + threadIdx.x;
  if (i >= total) return;
  int n = i % Dout; int t = i / Dout; int k = t % Din; int h = t / Din;
  dst[((size_t)h * Dout + n) * Din + k] = (_Float16)src[i];
}

__global__ void k_concat_f16(_Float16* __restrict__ dst,
                             const float* __restrict__ nt, const float* __restrict__ nl){
  long long i = (long long)blockIdx.x * blockDim.x + threadIdx.x;
  if (i >= (long long)NN * 64) return;
  int n = (int)(i >> 6), c = (int)(i & 63);
  float v = (c < 14) ? nt[(size_t)n * 14 + c] : nl[(size_t)n * 50 + (c - 14)];
  dst[i] = (_Float16)v;
}

// ---------------- WMMA GEMM: out[rows, NH*Dout] = A[rows,Din](f16) x Wt[h][Dout][Din](f16) + bias ----------------
__global__ __launch_bounds__(256)
void k_gemm_node(const _Float16* __restrict__ A, int rows, int Din,
                 const _Float16* __restrict__ Wt, const float* __restrict__ bias,
                 float* __restrict__ out, int Dout)
{
  const int lane  = threadIdx.x & 31;
  const int wave  = threadIdx.x >> 5;
  const int mbase = (blockIdx.x * 8 + wave) * 16;
  if (mbase >= rows) return;
  const int hh   = blockIdx.z;
  const int col  = blockIdx.y * 16 + (lane & 15);
  const int arow   = mbase + (lane & 15);
  const int kshift = (lane < 16) ? 0 : 8;    // A: K sub-groups per lane-half
  const int bkoff  = (lane < 16) ? 0 : 8;    // B: K sub-groups per lane-half

  const float bv = bias[hh * Dout + col];
  v8f c;
  #pragma unroll
  for (int r = 0; r < 8; ++r) c[r] = bv;

  const _Float16* ap = A + (size_t)arow * Din + kshift;
  const _Float16* bp = Wt + ((size_t)hh * Dout + col) * Din + bkoff;
  for (int kb = 0; kb < Din; kb += 32) {
    v16h a = cat16(*(const v8h*)(ap + kb), *(const v8h*)(ap + kb + 16));
    v16h b = cat16(*(const v8h*)(bp + kb), *(const v8h*)(bp + kb + 16));
    c = __builtin_amdgcn_wmma_f32_16x16x32_f16(false, a, false, b, (short)0, c, false, false);
  }
  const int rbase   = mbase + ((lane < 16) ? 0 : 8);
  const int ostride = NH * Dout;
  #pragma unroll
  for (int r = 0; r < 8; ++r)
    out[(size_t)(rbase + r) * ostride + hh * Dout + col] = c[r];
}

// ---------------- edge projection: Din=32, Dout=16, fused a_e-dot (pe) + optional leaky f16 output ----------------
__global__ __launch_bounds__(256)
void k_gemm_edge(const _Float16* __restrict__ A, int rows,
                 const _Float16* __restrict__ Wt, const float* __restrict__ bias,
                 const float* __restrict__ ae, float* __restrict__ pe,
                 _Float16* __restrict__ xeout)
{
  const int lane  = threadIdx.x & 31;
  const int wave  = threadIdx.x >> 5;
  const int mbase = (blockIdx.x * 8 + wave) * 16;
  if (mbase >= rows) return;
  const int hh  = blockIdx.y;
  const int col = lane & 15;
  const int arow   = mbase + col;
  const int kshift = (lane < 16) ? 0 : 8;
  const int bkoff  = (lane < 16) ? 0 : 8;

  const float bv = bias[hh * 16 + col];
  v8f c;
  #pragma unroll
  for (int r = 0; r < 8; ++r) c[r] = bv;

  const _Float16* ap = A + (size_t)arow * 32 + kshift;
  const _Float16* bp = Wt + ((size_t)hh * 16 + col) * 32 + bkoff;
  v16h a = cat16(*(const v8h*)(ap), *(const v8h*)(ap + 16));
  v16h b = cat16(*(const v8h*)(bp), *(const v8h*)(bp + 16));
  c = __builtin_amdgcn_wmma_f32_16x16x32_f16(false, a, false, b, (short)0, c, false, false);

  const int rbase = mbase + ((lane < 16) ? 0 : 8);
  const float aev = ae[hh * 16 + col];
  #pragma unroll
  for (int r = 0; r < 8; ++r){
    float t = c[r] * aev;            // pe = dot(a_e, ze) over the 16 cols of each lane-half
    t += __shfl_xor(t, 1);
    t += __shfl_xor(t, 2);
    t += __shfl_xor(t, 4);
    t += __shfl_xor(t, 8);
    if ((lane & 15) == 0) pe[(size_t)(rbase + r) * NH + hh] = t;
  }
  if (xeout){
    #pragma unroll
    for (int r = 0; r < 8; ++r)
      xeout[(size_t)(rbase + r) * (NH * 16) + hh * 16 + col] = (_Float16)leaky(c[r], 0.01f);
  }
}

// ---------------- per-node attention dots: ps = a_s.z, pd = a_d.z ----------------
__global__ void k_node_dots(const float* __restrict__ z, const float* __restrict__ as_,
                            const float* __restrict__ ad_, float* __restrict__ ps,
                            float* __restrict__ pd)
{
  int i = blockIdx.x * blockDim.x + threadIdx.x;
  if (i >= NN * NH) return;
  const int hh = i & (NH - 1);
  const float* zp = z + (size_t)i * 32;
  const float* av = as_ + hh * 32;
  const float* dv = ad_ + hh * 32;
  float s = 0.f, d = 0.f;
  #pragma unroll
  for (int k = 0; k < 32; ++k){ s += zp[k] * av[k]; d += zp[k] * dv[k]; }
  ps[i] = s; pd[i] = d;
}

// ---------------- edge softmax phases ----------------
__global__ void k_edge_logits(const float* __restrict__ ps, const float* __restrict__ pe,
                              const float* __restrict__ pd, const float* __restrict__ ab,
                              const int* __restrict__ src, const int* __restrict__ dst,
                              float* __restrict__ sx, float* __restrict__ m)
{
  long long i = (long long)blockIdx.x * blockDim.x + threadIdx.x;
  if (i >= (long long)NE * NH) return;
  const long long e = i >> 1; const int hh = (int)(i & 1);
  float s = ps[(size_t)src[e] * NH + hh] + pe[i] + pd[(size_t)dst[e] * NH + hh] + ab[hh];
  s = leaky(s, 0.2f);
  sx[i] = s;
  atomicMaxF(&m[(size_t)dst[e] * NH + hh], s);
}

__global__ void k_edge_exp(const int* __restrict__ dst, const float* __restrict__ m,
                           float* __restrict__ sx, float* __restrict__ den)
{
  long long i = (long long)blockIdx.x * blockDim.x + threadIdx.x;
  if (i >= (long long)NE * NH) return;
  const long long e = i >> 1; const int hh = (int)(i & 1);
  float mm = m[(size_t)dst[e] * NH + hh];
  if (!__builtin_isfinite(mm)) mm = 0.f;
  float ex = expf(sx[i] - mm);
  sx[i] = ex;
  atomicAdd(&den[(size_t)dst[e] * NH + hh], ex);
}

__global__ void k_edge_norm(const int* __restrict__ dst, const float* __restrict__ den,
                            float* __restrict__ sx)
{
  long long i = (long long)blockIdx.x * blockDim.x + threadIdx.x;
  if (i >= (long long)NE * NH) return;
  const long long e = i >> 1; const int hh = (int)(i & 1);
  sx[i] = sx[i] / (den[(size_t)dst[e] * NH + hh] + 1e-9f);
}

// ---------------- alpha-weighted scatter: h[dst] += alpha * v[src], [N, 128] ----------------
__global__ __launch_bounds__(256)
void k_aggregate(const float* __restrict__ alpha, const float* __restrict__ v,
                 const int* __restrict__ src, const int* __restrict__ dst,
                 float* __restrict__ hout)
{
  long long i = (long long)blockIdx.x * 256 + threadIdx.x;
  if (i >= (long long)NE * 128) return;
  const long long e = i >> 7;
  const int r  = (int)(i & 127);
  const int hh = r >> 6;
  const int s  = src[e], d = dst[e];
  const float* vp = v + (size_t)s * 128;
  __builtin_prefetch(vp + r, 0, 1);            // global_prefetch_b8 (gfx1250)
  const float a = alpha[(size_t)e * NH + hh];
  atomicAdd(&hout[(size_t)d * 128 + r], a * vp[r]);
}

__global__ void k_leaky_cast(const float* __restrict__ src, _Float16* __restrict__ dst, long long n){
  long long i = (long long)blockIdx.x * blockDim.x + threadIdx.x;
  if (i < n) dst[i] = (_Float16)leaky(src[i], 0.01f);
}

__global__ void k_graph_scatter(const float* __restrict__ hn, const int* __restrict__ gid,
                                float* __restrict__ gsum)
{
  long long i = (long long)blockIdx.x * blockDim.x + threadIdx.x;
  if (i >= (long long)NN * 128) return;
  const int n = (int)(i >> 7);
  atomicAdd(&gsum[(size_t)gid[n] * 128 + (i & 127)], hn[i]);
}

__global__ void k_fc(const float* __restrict__ gsum, const float* __restrict__ Wfc,
                     const float* __restrict__ bfc, float* __restrict__ out)
{
  const int t = threadIdx.x;           // 128 threads = 64 graphs x 2 classes
  const int g = t >> 1, j = t & 1;
  float s = bfc[j];
  #pragma unroll
  for (int k = 0; k < 128; ++k) s += gsum[(size_t)g * 128 + k] * Wfc[k * 2 + j];
  out[g * 2 + j] = s;
}

// ---------------- launcher ----------------
extern "C" void kernel_launch(void* const* d_in, const int* in_sizes, int n_in,
                              void* d_out, int out_size, void* d_ws, size_t ws_size,
                              hipStream_t stream) {
  const float* node_types  = (const float*)d_in[0];
  const float* node_labels = (const float*)d_in[1];
  const float* edge_labels = (const float*)d_in[2];
  const int*   src         = (const int*)d_in[3];
  const int*   dst         = (const int*)d_in[4];
  const int*   gid         = (const int*)d_in[5];
  const float* l0_Wn = (const float*)d_in[6];  const float* l0_bn = (const float*)d_in[7];
  const float* l0_Wv = (const float*)d_in[8];  const float* l0_bv = (const float*)d_in[9];
  const float* l0_We = (const float*)d_in[10]; const float* l0_be = (const float*)d_in[11];
  const float* l0_as = (const float*)d_in[12]; const float* l0_ae = (const float*)d_in[13];
  const float* l0_ad = (const float*)d_in[14]; const float* l0_ab = (const float*)d_in[15];
  const float* l1_Wn = (const float*)d_in[16]; const float* l1_bn = (const float*)d_in[17];
  const float* l1_Wv = (const float*)d_in[18]; const float* l1_bv = (const float*)d_in[19];
  const float* l1_We = (const float*)d_in[20]; const float* l1_be = (const float*)d_in[21];
  const float* l1_as = (const float*)d_in[22]; const float* l1_ae = (const float*)d_in[23];
  const float* l1_ad = (const float*)d_in[24]; const float* l1_ab = (const float*)d_in[25];
  const float* fc_W  = (const float*)d_in[26]; const float* fc_b  = (const float*)d_in[27];
  float* out = (float*)d_out;

  char* w = (char*)d_ws; size_t off = 0;
  auto take = [&](size_t bytes)->void* {
    void* p = w + off; off += (bytes + 255) & ~(size_t)255; return p;
  };
  _Float16* xn0h = (_Float16*)take((size_t)NN * 64 * 2);
  _Float16* xe0h = (_Float16*)take((size_t)NE * 32 * 2);
  _Float16* xe1h = (_Float16*)take((size_t)NE * 32 * 2);
  _Float16* wn0h = (_Float16*)take(2 * 64 * 32 * 2);
  _Float16* wv0h = (_Float16*)take(2 * 64 * 64 * 2);
  _Float16* we0h = (_Float16*)take(2 * 32 * 16 * 2);
  _Float16* wn1h = (_Float16*)take(2 * 128 * 32 * 2);
  _Float16* wv1h = (_Float16*)take(2 * 128 * 64 * 2);
  _Float16* we1h = (_Float16*)take(2 * 32 * 16 * 2);
  float* z    = (float*)take((size_t)NN * 64 * 4);
  float* v    = (float*)take((size_t)NN * 128 * 4);
  float* ps   = (float*)take((size_t)NN * NH * 4);
  float* pd   = (float*)take((size_t)NN * NH * 4);
  float* pe   = (float*)take((size_t)NE * NH * 4);
  float* sx   = (float*)take((size_t)NE * NH * 4);
  float* m    = (float*)take((size_t)NN * NH * 4);
  float* den  = (float*)take((size_t)NN * NH * 4);
  float* hbuf = (float*)take((size_t)NN * 128 * 4);
  _Float16* xn1h = (_Float16*)take((size_t)NN * 128 * 2);
  float* gsum = (float*)take((size_t)NG * 128 * 4);

  auto cdiv = [](long long a, long long b)->int { return (int)((a + b - 1) / b); };
  const float NEG_INF = -__builtin_huge_valf();

  // weight packs (f32 [H,Din,Dout] -> f16 [H,Dout,Din]) + input casts
  k_pack_w<<<cdiv(4096, 256),  256, 0, stream>>>(wn0h, l0_Wn, 64, 32, 4096);
  k_pack_w<<<cdiv(8192, 256),  256, 0, stream>>>(wv0h, l0_Wv, 64, 64, 8192);
  k_pack_w<<<cdiv(1024, 256),  256, 0, stream>>>(we0h, l0_We, 32, 16, 1024);
  k_pack_w<<<cdiv(8192, 256),  256, 0, stream>>>(wn1h, l1_Wn, 128, 32, 8192);
  k_pack_w<<<cdiv(16384, 256), 256, 0, stream>>>(wv1h, l1_Wv, 128, 64, 16384);
  k_pack_w<<<cdiv(1024, 256),  256, 0, stream>>>(we1h, l1_We, 32, 16, 1024);
  k_cast_f16<<<cdiv((long long)NE * 32, 256), 256, 0, stream>>>(xe0h, edge_labels, (long long)NE * 32);
  k_concat_f16<<<cdiv((long long)NN * 64, 256), 256, 0, stream>>>(xn0h, node_types, node_labels);

  const int nTilesB = cdiv(NN / 16, 8);   // 782
  const int eTilesB = cdiv(NE / 16, 8);   // 12500

  // ===== layer 0 =====
  k_gemm_node<<<dim3(nTilesB, 2, NH), 256, 0, stream>>>(xn0h, NN, 64, wn0h, l0_bn, z, 32);
  k_gemm_node<<<dim3(nTilesB, 4, NH), 256, 0, stream>>>(xn0h, NN, 64, wv0h, l0_bv, v, 64);
  k_gemm_edge<<<dim3(eTilesB, NH), 256, 0, stream>>>(xe0h, NE, we0h, l0_be, l0_ae, pe, xe1h);
  k_node_dots<<<cdiv((long long)NN * NH, 256), 256, 0, stream>>>(z, l0_as, l0_ad, ps, pd);
  k_fill_f32<<<cdiv((long long)NN * NH, 256), 256, 0, stream>>>(m, NEG_INF, (long long)NN * NH);
  k_fill_f32<<<cdiv((long long)NN * NH, 256), 256, 0, stream>>>(den, 0.f, (long long)NN * NH);
  k_fill_f32<<<cdiv((long long)NN * 128, 256), 256, 0, stream>>>(hbuf, 0.f, (long long)NN * 128);
  k_edge_logits<<<cdiv((long long)NE * NH, 256), 256, 0, stream>>>(ps, pe, pd, l0_ab, src, dst, sx, m);
  k_edge_exp<<<cdiv((long long)NE * NH, 256), 256, 0, stream>>>(dst, m, sx, den);
  k_edge_norm<<<cdiv((long long)NE * NH, 256), 256, 0, stream>>>(dst, den, sx);
  k_aggregate<<<cdiv((long long)NE * 128, 256), 256, 0, stream>>>(sx, v, src, dst, hbuf);
  k_leaky_cast<<<cdiv((long long)NN * 128, 256), 256, 0, stream>>>(hbuf, xn1h, (long long)NN * 128);

  // ===== layer 1 =====
  k_gemm_node<<<dim3(nTilesB, 2, NH), 256, 0, stream>>>(xn1h, NN, 128, wn1h, l1_bn, z, 32);
  k_gemm_node<<<dim3(nTilesB, 4, NH), 256, 0, stream>>>(xn1h, NN, 128, wv1h, l1_bv, v, 64);
  k_gemm_edge<<<dim3(eTilesB, NH), 256, 0, stream>>>(xe1h, NE, we1h, l1_be, l1_ae, pe, (_Float16*)nullptr);
  k_node_dots<<<cdiv((long long)NN * NH, 256), 256, 0, stream>>>(z, l1_as, l1_ad, ps, pd);
  k_fill_f32<<<cdiv((long long)NN * NH, 256), 256, 0, stream>>>(m, NEG_INF, (long long)NN * NH);
  k_fill_f32<<<cdiv((long long)NN * NH, 256), 256, 0, stream>>>(den, 0.f, (long long)NN * NH);
  k_fill_f32<<<cdiv((long long)NN * 128, 256), 256, 0, stream>>>(hbuf, 0.f, (long long)NN * 128);
  k_edge_logits<<<cdiv((long long)NE * NH, 256), 256, 0, stream>>>(ps, pe, pd, l1_ab, src, dst, sx, m);
  k_edge_exp<<<cdiv((long long)NE * NH, 256), 256, 0, stream>>>(dst, m, sx, den);
  k_edge_norm<<<cdiv((long long)NE * NH, 256), 256, 0, stream>>>(dst, den, sx);
  k_aggregate<<<cdiv((long long)NE * 128, 256), 256, 0, stream>>>(sx, v, src, dst, hbuf);

  // ===== readout =====
  k_fill_f32<<<cdiv((long long)NG * 128, 256), 256, 0, stream>>>(gsum, 0.f, (long long)NG * 128);
  k_graph_scatter<<<cdiv((long long)NN * 128, 256), 256, 0, stream>>>(hbuf, gid, gsum);
  k_fc<<<1, 128, 0, stream>>>(gsum, fc_W, fc_b, out);
}
